// OrbECG_72937134620845
// MI455X (gfx1250) — matrix operand order repaired and checked by reading, not compiled, not run on hardware
//
#include <hip/hip_runtime.h>
#include <hip/hip_bf16.h>
#include <hip/hip_fp16.h>

typedef __attribute__((ext_vector_type(16))) _Float16 v16h;
typedef __attribute__((ext_vector_type(8)))  float    v8f;

#define NATOM 16
#define NBOND 17
#define NPAIR 24
#define N1    33
#define BLK   81
#define NB    243
#define NBEAD 8
#define NENC  128

__device__ __constant__ int dE1[NBOND] = {0,0,1,2,2,3,4,5,6,8,8,9,10,11,11,13,14};
__device__ __constant__ int dE2[NBOND] = {1,4,2,3,5,4,12,6,7,9,12,10,11,12,13,14,15};
__device__ __constant__ int dP0[NPAIR] = {1,0,1,1,3,2,0,0,3,2,5,5,9,8,9,10,10,12,4,4,8,11,13,13};
__device__ __constant__ int dP1[NPAIR] = {4,2,3,5,5,4,3,12,12,6,7,6,12,10,11,12,13,13,8,11,11,14,15,14};
__device__ __constant__ int dNI[NPAIR] = {0,1,2,2,2,3,4,4,4,5,6,7,8,9,10,11,11,11,12,12,12,13,14,15};
__device__ __constant__ float dSIG[3]  = {0.4f, 0.8f, 1.2f};
__device__ __constant__ int dPA[6] = {0,0,0,1,1,2};
__device__ __constant__ int dPC[6] = {0,1,2,1,2,2};

// ---------------------------------------------------------------------------
// Kernel 0: convert MLP weights fp32 -> fp16 once into workspace.
// ---------------------------------------------------------------------------
__global__ void cvt_kernel(const float* __restrict__ w1, const float* __restrict__ w2,
                           const float* __restrict__ w3, _Float16* __restrict__ w16) {
    int i = blockIdx.x * 256 + threadIdx.x;     // 3 * 16384 total
    if (i < 16384)       w16[i] = (_Float16)w1[i];
    else if (i < 32768)  w16[i] = (_Float16)w2[i - 16384];
    else if (i < 49152)  w16[i] = (_Float16)w3[i - 32768];
}

// ---------------------------------------------------------------------------
// Kernel A: per batch item compute ovl = C * S * C^T  (8x8), f32 VALU.
// One workgroup (256 threads) per batch item; all data LDS-resident (~40 KB).
// Uses the sigma-block factorization: 81x81 d2 shared by all 9 sigma pairs.
// ---------------------------------------------------------------------------
__global__ __launch_bounds__(256) void ovl_kernel(const float* __restrict__ pos,
                                                  const float* __restrict__ cg,
                                                  const float* __restrict__ sigp,
                                                  float* __restrict__ ovl) {
    __shared__ float P[BLK][3];
    __shared__ float Cb[3][NBEAD][BLK];
    __shared__ float D2[BLK][BLK];
    __shared__ float U[2][NBEAD][BLK];
    __shared__ float sK[6], sB[6];

    const int t = threadIdx.x;
    const int b = blockIdx.x;
    const float* pb = pos + b * NATOM * 3;

    if (t < 6) {
        int a = dPA[t], c = dPC[t];
        float sa = dSIG[a] * expf(sigp[a]);
        float sc = dSIG[c] * expf(sigp[c]);
        float s2 = sa * sa + sc * sc;
        sB[t] = 1.0f / (2.0f * s2);
        sK[t] = powf(2.0f * sa * sc / s2, 1.5f);
    }
    if (t < BLK) {
        float x, y, z;
        if (t < NATOM) {
            x = pb[t*3]; y = pb[t*3+1]; z = pb[t*3+2];
        } else if (t < N1) {
            int e = t - NATOM; int i1 = dE1[e], i2 = dE2[e];
            x = 0.5f * (pb[i1*3+0] + pb[i2*3+0]);
            y = 0.5f * (pb[i1*3+1] + pb[i2*3+1]);
            z = 0.5f * (pb[i1*3+2] + pb[i2*3+2]);
        } else {
            int c = t - N1; int pr = c >> 1;
            float scl = (c & 1) ? 0.25f : -0.25f;
            int nc = dNI[pr], a0 = dP0[pr], a1 = dP1[pr];
            x = pb[nc*3+0] + scl * (pb[a0*3+0] - pb[a1*3+0]);
            y = pb[nc*3+1] + scl * (pb[a0*3+1] - pb[a1*3+1]);
            z = pb[nc*3+2] + scl * (pb[a0*3+2] - pb[a1*3+2]);
        }
        P[t][0] = x; P[t][1] = y; P[t][2] = z;
    }
    // C = make_opposite_blocks(cg_ref[0]) forward values
    for (int idx = t; idx < 3 * NBEAD * BLK; idx += 256) {
        int a = idx / (NBEAD * BLK);
        int r = idx % (NBEAD * BLK);
        int q = r / BLK, i = r % BLK;
        int col = a * BLK + i;
        float v;
        if (i < N1) v = cg[q * NB + col];
        else {
            int off = i - N1;
            v = (off & 1) ? -cg[q * NB + col - 1] : cg[q * NB + col];
        }
        Cb[a][q][i] = v;
    }
    __syncthreads();

    for (int idx = t; idx < BLK * BLK; idx += 256) {
        int i = idx / BLK, j = idx % BLK;
        float dx = P[i][0] - P[j][0];
        float dy = P[i][1] - P[j][1];
        float dz = P[i][2] - P[j][2];
        D2[i][j] = dx*dx + dy*dy + dz*dz;
    }
    __syncthreads();

    float acc = 0.0f;
    const int p = t >> 3, q = t & 7;
    for (int m = 0; m < 6; ++m) {
        int a = dPA[m], c = dPC[m];
        float beta = sB[m];
        if (t < BLK) {
            float uc[8] = {0,0,0,0,0,0,0,0};
            float ua[8] = {0,0,0,0,0,0,0,0};
            for (int j = 0; j < BLK; ++j) {
                float e = expf(-beta * D2[t][j]);
                #pragma unroll
                for (int qq = 0; qq < 8; ++qq) uc[qq] += e * Cb[c][qq][j];
                if (a != c) {
                    #pragma unroll
                    for (int qq = 0; qq < 8; ++qq) ua[qq] += e * Cb[a][qq][j];
                }
            }
            #pragma unroll
            for (int qq = 0; qq < 8; ++qq) {
                U[0][qq][t] = uc[qq];
                U[1][qq][t] = (a != c) ? ua[qq] : uc[qq];
            }
        }
        __syncthreads();
        if (t < 64) {
            float m1 = 0.0f, m2 = 0.0f;
            for (int i = 0; i < BLK; ++i) {
                m1 += Cb[a][p][i] * U[0][q][i];   // C_a E C_c^T
                m2 += Cb[c][p][i] * U[1][q][i];   // C_c E C_a^T
            }
            acc += (a == c) ? sK[m] * m1 : sK[m] * (m1 + m2);
        }
        __syncthreads();
    }
    if (t < 64) ovl[b * 64 + t] = acc;
}

// ---------------------------------------------------------------------------
// Kernel B: soft-binning softmax + 3x(128x128) MLP + softmax-projection.
// 128 threads = 4 waves; each wave owns a 16-row x 128-col tile.
// GEMMs via v_wmma_f32_16x16x32_f16 (4 K-steps x 8 N-tiles x 3 layers).
// ---------------------------------------------------------------------------
__global__ __launch_bounds__(128) void mlp_kernel(
    const float* __restrict__ xin,
    const float* __restrict__ encw, const float* __restrict__ encb,
    const float* __restrict__ mup, const float* __restrict__ lr,
    const _Float16* __restrict__ w16,
    const float* __restrict__ b1, const float* __restrict__ b2,
    const float* __restrict__ b3,
    float* __restrict__ out) {

    __shared__ __align__(16) _Float16 hbuf[4][16][NENC];   // activations (f16)
    __shared__ __align__(16) float    fbuf[4][16][NENC];   // final logits (f32)

    const int tid  = threadIdx.x;
    const int wv   = tid >> 5;
    const int lane = tid & 31;
    const int l16  = lane & 15;
    const int hh   = lane >> 4;
    const int rowBase = blockIdx.x * 64 + wv * 16;
    const int row  = rowBase + l16;      // lanes (l16, l16+16) share one row

    // ---- phase 1: e = softmax(enc_weight * x + enc_bias) -> hbuf (f16) ----
    {
        float x = xin[row];
        int f0 = hh * 64;
        float mx = -1e30f;
        for (int f = f0; f < f0 + 64; ++f) mx = fmaxf(mx, encw[f] * x + encb[f]);
        mx = fmaxf(mx, __shfl_xor(mx, 16));
        float sum = 0.0f;
        for (int f = f0; f < f0 + 64; ++f) {
            float s = expf(encw[f] * x + encb[f] - mx);
            sum += s;
            hbuf[wv][l16][f] = (_Float16)s;
        }
        sum += __shfl_xor(sum, 16);
        float inv = 1.0f / sum;
        for (int f = f0; f < f0 + 64; ++f)
            hbuf[wv][l16][f] = (_Float16)((float)hbuf[wv][l16][f] * inv);
    }
    __syncthreads();

    // ---- 3 dense layers, WMMA f16 -> f32 ----
    #pragma unroll
    for (int L = 0; L < 3; ++L) {
        const _Float16* W = w16 + L * 16384;                 // (out=N,in=K) row-major
        const float* bias = (L == 0) ? b1 : (L == 1) ? b2 : b3;

        union AF { unsigned u[8]; v16h v; };
        AF a[4];
        #pragma unroll
        for (int ks = 0; ks < 4; ++ks) {
            #pragma unroll
            for (int pp = 0; pp < 8; ++pp) {
                int k = ks * 32 + ((pp >> 2) << 4) + (hh << 3) + ((pp & 3) << 1);
                a[ks].u[pp] = *(const unsigned*)&hbuf[wv][l16][k];  // A: lane=M
            }
        }
        #pragma unroll
        for (int nt = 0; nt < 8; ++nt) {
            float bn = bias[nt * 16 + l16];
            v8f acc = {bn, bn, bn, bn, bn, bn, bn, bn};
            #pragma unroll
            for (int ks = 0; ks < 4; ++ks) {
                AF bf;
                #pragma unroll
                for (int pp = 0; pp < 8; ++pp) {
                    int k = ks * 32 + ((pp >> 2) << 4) + (hh << 3) + ((pp & 3) << 1);     // B: lane=N, packed K
                    bf.u[pp] = *(const unsigned*)&W[(nt * 16 + l16) * NENC + k];
                }
                acc = __builtin_amdgcn_wmma_f32_16x16x32_f16(
                        false, a[ks].v, false, bf.v, (short)0, acc, false, false);
            }
            #pragma unroll
            for (int r = 0; r < 8; ++r) {                        // D: M=r+8h, N=l16
                float v = acc[r];
                if (L < 2) {
                    v = fmaxf(v, 0.0f);
                    hbuf[wv][r + 8 * hh][nt * 16 + l16] = (_Float16)v;
                } else {
                    fbuf[wv][r + 8 * hh][nt * 16 + l16] = v;
                }
            }
        }
        __syncthreads();
    }

    // ---- final: softmax over 128 logits, dot with mu_proj, scale ----
    {
        int f0 = hh * 64;
        float mx = -1e30f;
        for (int f = f0; f < f0 + 64; ++f) mx = fmaxf(mx, fbuf[wv][l16][f]);
        mx = fmaxf(mx, __shfl_xor(mx, 16));
        float sum = 0.0f, dot = 0.0f;
        for (int f = f0; f < f0 + 64; ++f) {
            float s = expf(fbuf[wv][l16][f] - mx);
            sum += s;
            dot += s * mup[f];
        }
        sum += __shfl_xor(sum, 16);
        dot += __shfl_xor(dot, 16);
        if (hh == 0) out[row] = lr[0] * dot / sum;
    }
}

// ---------------------------------------------------------------------------
extern "C" void kernel_launch(void* const* d_in, const int* in_sizes, int n_in,
                              void* d_out, int out_size, void* d_ws, size_t ws_size,
                              hipStream_t stream) {
    const float* pos  = (const float*)d_in[0];
    const float* cg   = (const float*)d_in[1];
    const float* sigp = (const float*)d_in[2];
    const float* encw = (const float*)d_in[3];
    const float* encb = (const float*)d_in[4];
    const float* mup  = (const float*)d_in[5];
    const float* lr   = (const float*)d_in[6];
    const float* W1   = (const float*)d_in[7];
    const float* b1   = (const float*)d_in[8];
    const float* W2   = (const float*)d_in[9];
    const float* b2   = (const float*)d_in[10];
    const float* W3   = (const float*)d_in[11];
    const float* b3   = (const float*)d_in[12];
    float* out = (float*)d_out;

    char* ws = (char*)d_ws;
    float*     ovl = (float*)ws;                              // 65536 f32 (256 KB)
    _Float16*  w16 = (_Float16*)(ws + 65536 * sizeof(float)); // 3*16384 f16 (96 KB)

    cvt_kernel<<<192, 256, 0, stream>>>(W1, W2, W3, w16);
    ovl_kernel<<<1024, 256, 0, stream>>>(pos, cg, sigp, ovl);
    mlp_kernel<<<1024, 128, 0, stream>>>(ovl, encw, encb, mup, lr, w16,
                                         b1, b2, b3, out);
}
